// Pointcnn_cls_feature_37881611551318
// MI455X (gfx1250) — matrix operand order repaired and checked by reading, not compiled, not run on hardware
//
#include <hip/hip_runtime.h>
#include <hip/hip_bf16.h>
#include <cstddef>
#include <cstdint>

// ---------------------------------------------------------------------------
// PointCNN feature extractor for MI455X (gfx1250, wave32, WMMA + async LDS).
//   pc (16,2048,3) f32 ; params per LAYER_CFG ; out (16,128,480) f32
// pts at every layer is a prefix of pc, so only feature buffers live in ws.
// ---------------------------------------------------------------------------

typedef __attribute__((ext_vector_type(16))) _Float16 v16h;
typedef __attribute__((ext_vector_type(8)))  _Float16 v8h;
typedef __attribute__((ext_vector_type(8)))  float    v8f;
typedef __attribute__((ext_vector_type(4)))  float    v4f;
typedef __attribute__((ext_vector_type(4)))  _Float16 v4h;

#define NB   16          // batch
#define NPC  2048        // points per cloud
#define NPC3 (NPC * 3)

__device__ __forceinline__ float elu1(float x) {
    return x > 0.0f ? x : (expf(x) - 1.0f);
}

// 16-byte async copy global -> LDS (ASYNCcnt-tracked, per-lane addresses)
__device__ __forceinline__ void async_copy16(unsigned ldsDst, const void* src) {
    asm volatile("global_load_async_to_lds_b128 %0, %1, off"
                 :: "v"(ldsDst),
                    "v"((unsigned long long)(uintptr_t)src)
                 : "memory");
}
__device__ __forceinline__ void async_wait0() {
    asm volatile("s_wait_asynccnt 0x0" ::: "memory");
}

__device__ __forceinline__ v16h cat8(v8h lo, v8h hi) {
    v16h r;
#pragma unroll
    for (int h = 0; h < 8; ++h) { r[h] = lo[h]; r[h + 8] = hi[h]; }
    return r;
}

// ---------------------------------------------------------------------------
// 1) Brute-force dilated KNN.  One thread per query point, LDS-tiled over the
//    candidate points.  Sorted candidate list lives entirely in VGPRs: the
//    insert is a fully unrolled bubble pass (constant indices -> no scratch).
// ---------------------------------------------------------------------------
template <int K, int D>
__global__ __launch_bounds__(128)
void knn_kernel(const float* __restrict__ pc, int N, int P,
                int* __restrict__ idx_out) {
    constexpr int KD = K * D;
    constexpr int TS = 128;
    __shared__ float s_pts[TS * 3];

    const int blocksPerBatch = P / TS;
    const int b = blockIdx.x / blocksPerBatch;
    const int p = (blockIdx.x % blocksPerBatch) * TS + threadIdx.x;
    const float* base = pc + (size_t)b * NPC3;

    const float qx = base[p * 3 + 0];
    const float qy = base[p * 3 + 1];
    const float qz = base[p * 3 + 2];

    float bd[KD];
    int   bi[KD];
#pragma unroll
    for (int i = 0; i < KD; ++i) { bd[i] = 3.0e38f; bi[i] = 0; }

    for (int n0 = 0; n0 < N; n0 += TS) {
        __syncthreads();
        const int lim = (N - n0 < TS) ? (N - n0) : TS;
        for (int e = threadIdx.x; e < lim * 3; e += TS)
            s_pts[e] = base[n0 * 3 + e];
        if (n0 + TS < N)
            __builtin_prefetch(base + (size_t)(n0 + TS) * 3, 0, 3);
        __syncthreads();

        for (int nn = 0; nn < lim; ++nn) {
            const float dx = s_pts[nn * 3 + 0] - qx;
            const float dy = s_pts[nn * 3 + 1] - qy;
            const float dz = s_pts[nn * 3 + 2] - qz;
            const float d2 = dx * dx + dy * dy + dz * dz;
            if (d2 < bd[KD - 1]) {          // rare path: register bubble insert
                bd[KD - 1] = d2;
                bi[KD - 1] = n0 + nn;
#pragma unroll
                for (int j = KD - 1; j > 0; --j) {
                    // strict < keeps lower index first on distance ties
                    if (bd[j] < bd[j - 1]) {
                        const float td = bd[j]; bd[j] = bd[j - 1]; bd[j - 1] = td;
                        const int   ti = bi[j]; bi[j] = bi[j - 1]; bi[j - 1] = ti;
                    }
                }
            }
        }
    }

    int* op = idx_out + (size_t)(b * P + p) * K;
#pragma unroll
    for (int k = 0; k < K; ++k) op[k] = bi[k * D];   // idx[:, :, ::D]
}

// ---------------------------------------------------------------------------
// 2) X-conv feature stage.  One wave (32 lanes) per query point.  Small
//    MLPs / X-transform einsums on VALU through LDS; fX = Xm @ f on
//    v_wmma_f32_16x16x32_f16 with zero-padded operands.  f is kept transposed
//    in LDS (s_fT[c][j]) so both WMMA operands load as two ds_load_b128.
//    Output dwflat is f16 with row padded to a multiple of 32 so the GEMM's
//    A tile is a pure byte copy (async global->LDS).
// ---------------------------------------------------------------------------
struct FeatParams {
    const float* d1_w; const float* d1_b;
    const float* d2_w; const float* d2_b;
    const float* xt0_w; const float* xt0_b;
    const float* xt1_w; const float* xt2_w;
    const float* dw_w;
};

template <int K, int Cd, int Cin, int MULT>
__global__ __launch_bounds__(32)
void xconv_feat_kernel(const float* __restrict__ pc,
                       const float* __restrict__ fts_in,
                       const int* __restrict__ idx,
                       FeatParams fp,
                       _Float16* __restrict__ dwflat,
                       int P, int ftsBatchStride /* Nf*Cin */) {
    constexpr int Cmid  = Cd + Cin;
    constexpr int CMIDP = (Cmid + 15) & ~15;
    constexpr int CmM   = Cmid * MULT;
    constexpr int CmMP  = (CmM + 31) & ~31;   // padded f16 row for GEMM A
    constexpr int K2    = K * K;

    __shared__ float    s_local[16][3];
    __shared__ int      s_idx[16];
    __shared__ float    s_f1[16][Cd];
    __shared__ _Float16 s_fT[CMIDP][32];    // f transposed: [c][j], zero pad
    __shared__ float    s_xa[256];
    __shared__ float    s_xb[256];
    __shared__ _Float16 s_Xm[16][32];       // final X-transform, padded
    __shared__ float    s_fx[16][CMIDP];    // fX result

    const int lane = threadIdx.x;
    const int b = blockIdx.x / P;
    const int p = blockIdx.x % P;
    const float* pbase = pc + (size_t)b * NPC3;
    const float qx = pbase[p * 3 + 0];
    const float qy = pbase[p * 3 + 1];
    const float qz = pbase[p * 3 + 2];

    // zero f (covers K..31 rows and Cmid..CMIDP cols used by WMMA padding)
    for (int e = lane; e < CMIDP * 32; e += 32)
        (&s_fT[0][0])[e] = (_Float16)0.0f;

    if (lane < K) {
        const int i = idx[(size_t)(b * P + p) * K + lane];
        s_idx[lane] = i;
        s_local[lane][0] = pbase[i * 3 + 0] - qx;
        s_local[lane][1] = pbase[i * 3 + 1] - qy;
        s_local[lane][2] = pbase[i * 3 + 2] - qz;
    }
    __syncthreads();

    // f1 = elu(local @ d1_w + d1_b)
    for (int e = lane; e < K * Cd; e += 32) {
        const int k = e / Cd, c = e % Cd;
        float acc = fp.d1_b[c];
#pragma unroll
        for (int d = 0; d < 3; ++d) acc += s_local[k][d] * fp.d1_w[d * Cd + c];
        s_f1[k][c] = elu1(acc);
    }
    __syncthreads();

    // f2 = elu(f1 @ d2_w + d2_b)  ->  s_fT[0:Cd][k]
    for (int e = lane; e < K * Cd; e += 32) {
        const int k = e / Cd, c = e % Cd;
        float acc = fp.d2_b[c];
        for (int j = 0; j < Cd; ++j) acc += s_f1[k][j] * fp.d2_w[j * Cd + c];
        s_fT[c][k] = (_Float16)elu1(acc);
    }
    // gather previous features -> s_fT[Cd:Cd+Cin][k]  (b128 loads, Cin%4==0)
    if (Cin > 0) {
        constexpr int C4 = (Cin > 0) ? (Cin / 4) : 1;
        for (int e = lane; e < K * C4; e += 32) {
            const int k = e / C4, c4 = (e % C4) * 4;
            const v4f t = *(const v4f*)(fts_in + (size_t)b * ftsBatchStride +
                                        (size_t)s_idx[k] * Cin + c4);
#pragma unroll
            for (int i = 0; i < 4; ++i)
                s_fT[Cd + c4 + i][k] = (_Float16)t[i];
        }
    }

    // Xm0 = elu(einsum('kd,kdm->m', local, xt0_w) + xt0_b)
    for (int m = lane; m < K2; m += 32) {
        float acc = fp.xt0_b[m];
#pragma unroll
        for (int k = 0; k < K; ++k)
#pragma unroll
            for (int d = 0; d < 3; ++d)
                acc += s_local[k][d] * fp.xt0_w[(k * 3 + d) * K2 + m];
        s_xa[m] = elu1(acc);
    }
    __syncthreads();

    // Xm1[c,m] = elu(sum_k Xm0[k,c] * xt1_w[k,c,m])
    for (int e = lane; e < K2; e += 32) {
        const int c = e / K, m = e % K;
        float acc = 0.0f;
        for (int k = 0; k < K; ++k)
            acc += s_xa[k * K + c] * fp.xt1_w[(k * K + c) * K + m];
        s_xb[c * K + m] = elu1(acc);
    }
    __syncthreads();

    // Xm2[c,m] = sum_k Xm1[k,c] * xt2_w[k,c,m]
    for (int e = lane; e < K2; e += 32) {
        const int c = e / K, m = e % K;
        float acc = 0.0f;
        for (int k = 0; k < K; ++k)
            acc += s_xb[k * K + c] * fp.xt2_w[(k * K + c) * K + m];
        s_xa[c * K + m] = acc;
    }
    __syncthreads();

    // pad Xm2 into 16x32 f16 tile (A operand)
    for (int e = lane; e < 16 * 32; e += 32) {
        const int r = e >> 5, cc = e & 31;
        s_Xm[r][cc] = (r < K && cc < K) ? (_Float16)s_xa[r * K + cc]
                                        : (_Float16)0.0f;
    }
    __syncthreads();

    // fX = Xm2 @ f  via WMMA (contracted dim zero-padded to 32).
    // A halves: s_Xm[r][abase + 0..7] and s_Xm[r][abase+16 + 0..7]
    // B halves: s_fT[n0+r][bbase + 0..7] and s_fT[n0+r][bbase+16 + 0..7]  (two b128 loads each)
    {
        const int r     = lane & 15;
        const int abase = (lane >= 16) ? 8 : 0;
        const int bbase = (lane >= 16) ? 8 : 0;
        for (int n0 = 0; n0 < CMIDP; n0 += 16) {
            const v16h a  = cat8(*(const v8h*)&s_Xm[r][abase],
                                 *(const v8h*)&s_Xm[r][abase + 16]);
            const v16h bv = cat8(*(const v8h*)&s_fT[n0 + r][bbase],
                                 *(const v8h*)&s_fT[n0 + r][bbase + 16]);
            v8f acc = {};
            acc = __builtin_amdgcn_wmma_f32_16x16x32_f16(
                false, a, false, bv, (short)0, acc, false, false);
#pragma unroll
            for (int v = 0; v < 8; ++v) s_fx[v + abase][n0 + r] = acc[v];
        }
    }
    __syncthreads();

    // dw[c,m] = sum_k fX[k,c] * dw_w[k,c,m]  ->  dwflat row (c*MULT + m), f16
    _Float16* op = dwflat + (size_t)(b * P + p) * CmMP;
    for (int e = lane; e < CmMP; e += 32) {
        if (e < CmM) {
            const int c = e / MULT, mm = e % MULT;
            float acc = 0.0f;
            for (int k = 0; k < K; ++k)
                acc += s_fx[k][c] * fp.dw_w[(k * Cmid + c) * MULT + mm];
            op[e] = (_Float16)acc;
        } else {
            op[e] = (_Float16)0.0f;     // K-dim zero padding for the GEMM
        }
    }
}

// ---------------------------------------------------------------------------
// 3a) Weight prep: WpT[c][k] = f16(W[k][c]), k zero-padded to KdP.
//     Makes the GEMM's W tile a contiguous byte copy (async global->LDS).
// ---------------------------------------------------------------------------
__global__ void wprep_kernel(const float* __restrict__ W,
                             _Float16* __restrict__ WpT,
                             int N, int Kd, int KdP) {
    const int t = blockIdx.x * blockDim.x + threadIdx.x;
    if (t >= N * KdP) return;
    const int c = t / KdP, k = t % KdP;
    WpT[t] = (k < Kd) ? (_Float16)W[(size_t)k * N + c] : (_Float16)0.0f;
}

// ---------------------------------------------------------------------------
// 3b) Pointwise GEMM: out = elu(dwflat(M x KdP f16) @ W(KdP x N f16) + bias).
//     One wave per 16x16 tile.  Both tiles are pre-padded f16 and copied
//     global->LDS with global_load_async_to_lds_b128 (no branches, no cvt in
//     the hot loop); operands load as two ds_load_b128 each; MAC on
//     v_wmma_f32_16x16x32_f16.
// ---------------------------------------------------------------------------
__global__ __launch_bounds__(32)
void pw_gemm_kernel(const _Float16* __restrict__ A,
                    const _Float16* __restrict__ WpT,
                    const float* __restrict__ bias, float* __restrict__ out,
                    int KdP, int outStride, int outColOff, int tilesN) {
    __shared__ _Float16 s_A[16][32];     // [row][k]
    __shared__ _Float16 s_Wt[16][32];    // [col][k]

    const int tile = blockIdx.x;
    const int row0 = (tile / tilesN) * 16;
    const int col0 = (tile % tilesN) * 16;
    const int lane = threadIdx.x;
    const unsigned ldsA = (unsigned)(uintptr_t)(&s_A[0][0]);
    const unsigned ldsW = (unsigned)(uintptr_t)(&s_Wt[0][0]);

    const int rA = lane >> 2, qA = lane & 3;     // 4 chunks per 64B row

    v8f acc = {};
    for (int k0 = 0; k0 < KdP; k0 += 32) {
        __syncthreads();
        // A tile rows 0..15, W tile cols 0..15: 64 x 16B chunks each,
        // issued as 2 explicit per-lane async copies (no loop, no branches)
        async_copy16(ldsA + (unsigned)(rA * 64 + qA * 16),
                     A + (size_t)(row0 + rA) * KdP + k0 + qA * 8);
        async_copy16(ldsA + (unsigned)((rA + 8) * 64 + qA * 16),
                     A + (size_t)(row0 + rA + 8) * KdP + k0 + qA * 8);
        async_copy16(ldsW + (unsigned)(rA * 64 + qA * 16),
                     WpT + (size_t)(col0 + rA) * KdP + k0 + qA * 8);
        async_copy16(ldsW + (unsigned)((rA + 8) * 64 + qA * 16),
                     WpT + (size_t)(col0 + rA + 8) * KdP + k0 + qA * 8);
        async_wait0();
        __syncthreads();

        const int r     = lane & 15;
        const int abase = (lane >= 16) ? 8 : 0;
        const int bbase = (lane >= 16) ? 8 : 0;
        const v16h a  = cat8(*(const v8h*)&s_A[r][abase],
                             *(const v8h*)&s_A[r][abase + 16]);
        const v16h bv = cat8(*(const v8h*)&s_Wt[r][bbase],
                             *(const v8h*)&s_Wt[r][bbase + 16]);
        acc = __builtin_amdgcn_wmma_f32_16x16x32_f16(
            false, a, false, bv, (short)0, acc, false, false);
    }

    const int c  = col0 + (lane & 15);
    const int rb = row0 + ((lane >= 16) ? 8 : 0);
    const float bs = bias[c];
#pragma unroll
    for (int v = 0; v < 8; ++v) {
        const float val = elu1(acc[v] + bs);
        out[(size_t)(rb + v) * outStride + outColOff + c] = val;
    }
}

// ---------------------------------------------------------------------------
// 4) Layer-3 g-branch (two tiny dense layers on the query coordinates),
//    writes d_out columns [0,96) with row stride 480.
// ---------------------------------------------------------------------------
__global__ void g1_kernel(const float* __restrict__ pc,
                          const float* __restrict__ w,
                          const float* __restrict__ bias,
                          float* __restrict__ out, int P) {
    const int t = blockIdx.x * blockDim.x + threadIdx.x;
    const int total = NB * P * 96;
    if (t >= total) return;
    const int c = t % 96;
    const int bp = t / 96;
    const int b = bp / P, p = bp % P;
    const float* q = pc + (size_t)b * NPC3 + (size_t)p * 3;
    float acc = bias[c];
#pragma unroll
    for (int d = 0; d < 3; ++d) acc += q[d] * w[d * 96 + c];
    out[t] = elu1(acc);
}

__global__ void g2_kernel(const float* __restrict__ tmp,
                          const float* __restrict__ w,
                          const float* __restrict__ bias,
                          float* __restrict__ out, int P) {
    const int t = blockIdx.x * blockDim.x + threadIdx.x;
    const int total = NB * P * 96;
    if (t >= total) return;
    const int c = t % 96;
    const int bp = t / 96;
    const float* tr = tmp + (size_t)bp * 96;
    float acc = bias[c];
    for (int j = 0; j < 96; ++j) acc += tr[j] * w[j * 96 + c];
    out[(size_t)bp * 480 + c] = elu1(acc);
}

// ---------------------------------------------------------------------------
// Host launcher.  Input flattening assumes JAX pytree order: pc first, then
// each layer dict with keys sorted alphabetically:
//   L0..L2 (11): d1_b d1_w d2_b d2_w dw_w pw_b pw_w xt0_b xt0_w xt1_w xt2_w
//   L3     (15): d1_b d1_w d2_b d2_w dw_w g1_b g1_w g2_b g2_w pw_b pw_w
//                xt0_b xt0_w xt1_w xt2_w
// ---------------------------------------------------------------------------
extern "C" void kernel_launch(void* const* d_in, const int* in_sizes, int n_in,
                              void* d_out, int out_size, void* d_ws,
                              size_t ws_size, hipStream_t stream) {
    (void)in_sizes; (void)n_in; (void)out_size; (void)ws_size;
    const float* pc = (const float*)d_in[0];
    auto F = [&](int i) { return (const float*)d_in[i]; };

    // workspace layout
    float*     ws   = (float*)d_ws;
    float*     fts1 = ws;                        // 16*2048*48  = 1572864 f32
    float*     fts2 = fts1 + 1572864;            // 16*384*96   =  589824 f32
    float*     fts3 = fts2 + 589824;             // 16*128*192  =  393216 f32
    float*     gtmp = fts3 + 393216;             // 16*128*96   =  196608 f32
    _Float16*  dwf  = (_Float16*)(gtmp + 196608);// max 32768*96 = 3145728 f16
    int*       idxb = (int*)(dwf + 3145728);     // max 16*2048*8 = 262144 i32
    _Float16*  wpT  = (_Float16*)(idxb + 262144);// max 384*480  =  184320 f16

    // ---------------- Layer 0: K=8 D=1 P=2048 Cd=24 Cin=0 mult=4 Co=48
    knn_kernel<8, 1><<<NB * (2048 / 128), 128, 0, stream>>>(pc, 2048, 2048, idxb);
    {
        FeatParams fp{F(2), F(1), F(4), F(3), F(9), F(8), F(10), F(11), F(5)};
        xconv_feat_kernel<8, 24, 0, 4>
            <<<NB * 2048, 32, 0, stream>>>(pc, nullptr, idxb, fp, dwf, 2048, 0);
    }
    wprep_kernel<<<(48 * 96 + 255) / 256, 256, 0, stream>>>(F(7), wpT, 48, 96, 96);
    pw_gemm_kernel<<<(32768 / 16) * (48 / 16), 32, 0, stream>>>(
        dwf, wpT, F(6), fts1, 96, 48, 0, 48 / 16);

    // ---------------- Layer 1: K=12 D=2 P=384 Cd=12 Cin=48 mult=2 Co=96
    knn_kernel<12, 2><<<NB * (384 / 128), 128, 0, stream>>>(pc, 2048, 384, idxb);
    {
        FeatParams fp{F(13), F(12), F(15), F(14), F(20), F(19), F(21), F(22), F(16)};
        xconv_feat_kernel<12, 12, 48, 2>
            <<<NB * 384, 32, 0, stream>>>(pc, fts1, idxb, fp, dwf, 384, 2048 * 48);
    }
    wprep_kernel<<<(96 * 128 + 255) / 256, 256, 0, stream>>>(F(18), wpT, 96, 120, 128);
    pw_gemm_kernel<<<(6144 / 16) * (96 / 16), 32, 0, stream>>>(
        dwf, wpT, F(17), fts2, 128, 96, 0, 96 / 16);

    // ---------------- Layer 2: K=16 D=2 P=128 Cd=24 Cin=96 mult=2 Co=192
    knn_kernel<16, 2><<<NB * (128 / 128), 128, 0, stream>>>(pc, 384, 128, idxb);
    {
        FeatParams fp{F(24), F(23), F(26), F(25), F(31), F(30), F(32), F(33), F(27)};
        xconv_feat_kernel<16, 24, 96, 2>
            <<<NB * 128, 32, 0, stream>>>(pc, fts2, idxb, fp, dwf, 128, 384 * 96);
    }
    wprep_kernel<<<(192 * 256 + 255) / 256, 256, 0, stream>>>(F(29), wpT, 192, 240, 256);
    pw_gemm_kernel<<<(2048 / 16) * (192 / 16), 32, 0, stream>>>(
        dwf, wpT, F(28), fts3, 256, 192, 0, 192 / 16);

    // ---------------- Layer 3: K=16 D=3 P=128 Cd=48 Cin=192 mult=2 Co=384 +g
    knn_kernel<16, 3><<<NB * (128 / 128), 128, 0, stream>>>(pc, 128, 128, idxb);
    {
        FeatParams fp{F(35), F(34), F(37), F(36), F(46), F(45), F(47), F(48), F(38)};
        xconv_feat_kernel<16, 48, 192, 2>
            <<<NB * 128, 32, 0, stream>>>(pc, fts3, idxb, fp, dwf, 128, 128 * 192);
    }
    wprep_kernel<<<(384 * 480 + 255) / 256, 256, 0, stream>>>(F(44), wpT, 384, 480, 480);
    // pw output -> d_out columns [96,480), row stride 480
    pw_gemm_kernel<<<(2048 / 16) * (384 / 16), 32, 0, stream>>>(
        dwf, wpT, F(43), (float*)d_out, 480, 480, 96, 384 / 16);
    // g branch -> d_out columns [0,96)
    g1_kernel<<<(NB * 128 * 96 + 255) / 256, 256, 0, stream>>>(
        pc, F(40), F(39), gtmp, 128);
    g2_kernel<<<(NB * 128 * 96 + 255) / 256, 256, 0, stream>>>(
        gtmp, F(42), F(41), (float*)d_out, 128);
}
